// RecurrentEncoder_42494406426972
// MI455X (gfx1250) — compile-verified, matrix operand order checked by reading, not run on hardware
//
#include <hip/hip_runtime.h>
#include <stdint.h>

#define DEVINL __device__ __forceinline__

typedef __bf16 bf16_t;
typedef __attribute__((ext_vector_type(16))) __bf16 v16bf;
typedef __attribute__((ext_vector_type(8)))  __bf16 v8bf;
typedef __attribute__((ext_vector_type(8)))  float   v8f;

constexpr int kB = 256, kT = 64, kIN = 512, kH = 512, kC = 4, kG = 3 * kH; // kG=1536
constexpr int nWemb = kC * kH * kIN;               // 1,048,576
constexpr int nWg   = kC * kG * kH;                // 3,145,728
constexpr int nX    = kB * kT * kIN;               // 8,388,608
constexpr long long nXG = (long long)kB * kT * kG; // 25,165,824 floats
constexpr int MAX_TILES = 24;

DEVINL v8f zero8() { v8f z = {0.f,0.f,0.f,0.f,0.f,0.f,0.f,0.f}; return z; }

// Lane L owns row (L%16); K-chunks at kb..kb+7 and kb+16..kb+23 where kb=(L/16)*8 (+32*kstep).
// Matches the CDNA5 16-bit A 16x32 layout; B 32x16 is the mirrored layout, so B fragments
// load rows of the (row-major) weight matrix with identical addressing.
DEVINL v16bf load_frag(const bf16_t* base, int row, int ldk, int kb) {
  const v8bf c0 = *(const v8bf*)(base + row * ldk + kb);
  const v8bf c1 = *(const v8bf*)(base + row * ldk + kb + 16);
  v16bf r;
#pragma unroll
  for (int i = 0; i < 8; ++i) { r[i] = c0[i]; r[8 + i] = c1[i]; }
  return r;
}

// Error-compensated bf16x3 product: ~fp32 accuracy, f32 accumulate in the WMMA.
DEVINL v8f wmma3(v16bf ahi, v16bf alo, v16bf bhi, v16bf blo, v8f acc) {
  acc = __builtin_amdgcn_wmma_f32_16x16x32_bf16(false, ahi, false, bhi, (short)0, acc, false, false);
  acc = __builtin_amdgcn_wmma_f32_16x16x32_bf16(false, ahi, false, blo, (short)0, acc, false, false);
  acc = __builtin_amdgcn_wmma_f32_16x16x32_bf16(false, alo, false, bhi, (short)0, acc, false, false);
  return acc;
}

// CDNA5 async global->LDS copy (ASYNCcnt-tracked), per CDNA5 ISA.
// vdst VGPR carries the wave-relative LDS byte address; GV mode 64-bit vaddr.
DEVINL void async_copy_b128(uint32_t lds_byte_off, const void* gaddr) {
  asm volatile("global_load_async_to_lds_b128 %0, %1, off"
               :: "v"(lds_byte_off), "v"((uint64_t)(uintptr_t)gaddr)
               : "memory");
}
DEVINL void wait_async0() { asm volatile("s_wait_asynccnt 0x0" ::: "memory"); }

// ---------------- Kernel 1: split fp32 -> bf16 hi/lo ----------------
__global__ void split_kernel(const float* __restrict__ src, bf16_t* __restrict__ hi,
                             bf16_t* __restrict__ lo, int n) {
  int i = blockIdx.x * blockDim.x + threadIdx.x;
  if (i < n) {
    float f = src[i];
    __bf16 h = (__bf16)f;
    hi[i] = h;
    lo[i] = (__bf16)(f - (float)h);
  }
}

// ---------------- Kernel 2: group samples by camera into 16-row tiles ----------------
__global__ void build_tiles_kernel(const int* __restrict__ cam, int* __restrict__ tile_cam,
                                   int* __restrict__ tile_rows, int* __restrict__ n_tiles) {
  if (threadIdx.x != 0 || blockIdx.x != 0) return;
  int nt = 0;
  for (int c = 0; c < kC; ++c) {
    int cnt = 0, cur = -1;
    for (int b = 0; b < kB; ++b) {
      if (cam[b] == c) {
        if ((cnt & 15) == 0) {
          cur = nt++;
          tile_cam[cur] = c;
          for (int m = 0; m < 16; ++m) tile_rows[cur * 16 + m] = -1;
        }
        tile_rows[cur * 16 + (cnt & 15)] = b;
        ++cnt;
      }
    }
  }
  *n_tiles = nt;
}

// ---------------- Kernel 3: fused emb + xg precompute ----------------
// WG = (sample b, 16 t-rows). Async-stage the x tile (bf16 hi/lo) into LDS once,
// E = x@W_emb[c]^T + b_emb -> same LDS buffers, XG = E@W_ih[c]^T + b_ih -> ws (fp32).
__global__ __launch_bounds__(256) void emb_xg_kernel(
    const bf16_t* __restrict__ x_hi, const bf16_t* __restrict__ x_lo,
    const int* __restrict__ cam,
    const bf16_t* __restrict__ Wemb_hi, const bf16_t* __restrict__ Wemb_lo,
    const float* __restrict__ b_emb,
    const bf16_t* __restrict__ Wih_hi, const bf16_t* __restrict__ Wih_lo,
    const float* __restrict__ b_ih,
    float* __restrict__ xg) {
  // Phase 1: holds the 16x512 x tile (hi/lo). Phase 2: reused for the E tile.
  __shared__ __align__(16) bf16_t Shi[16 * kH];
  __shared__ __align__(16) bf16_t Slo[16 * kH];

  const int blk = blockIdx.x;
  const int b = blk >> 2;
  const int t0 = (blk & 3) * 16;
  const int w = threadIdx.x >> 5;
  const int lane = threadIdx.x & 31;
  const int half = lane >> 4;
  const int l15 = lane & 15;
  const int c = cam[b];

  // ---- async stage of the contiguous 16x512 bf16 tile (16KB per buffer) ----
  {
    const size_t tbase = (size_t)(b * kT + t0) * kIN; // rows are consecutive -> contiguous
    const char* gh = (const char*)(x_hi + tbase) + threadIdx.x * 64;
    const char* gl = (const char*)(x_lo + tbase) + threadIdx.x * 64;
    const uint32_t lh = (uint32_t)(uintptr_t)Shi + threadIdx.x * 64;
    const uint32_t ll = (uint32_t)(uintptr_t)Slo + threadIdx.x * 64;
#pragma unroll
    for (int q = 0; q < 4; ++q) {
      async_copy_b128(lh + q * 16, gh + q * 16);
      async_copy_b128(ll + q * 16, gl + q * 16);
    }
    wait_async0();
    __syncthreads();
  }

  const bf16_t* WEh = Wemb_hi + (size_t)c * kH * kIN;
  const bf16_t* WEl = Wemb_lo + (size_t)c * kH * kIN;
  const bf16_t* WIh = Wih_hi + (size_t)c * kG * kH;
  const bf16_t* WIl = Wih_lo + (size_t)c * kG * kH;

  // ---- GEMM1: E (16 x 512), 32 col-tiles, 4 per wave; A from LDS ----
  v8f acc[4];
#pragma unroll
  for (int i = 0; i < 4; ++i) acc[i] = zero8();
  for (int k = 0; k < kIN / 32; ++k) {
    const int kb = half * 8 + k * 32;
    v16bf ahi = load_frag(Shi, l15, kIN, kb);
    v16bf alo = load_frag(Slo, l15, kIN, kb);
#pragma unroll
    for (int i = 0; i < 4; ++i) {
      const int nrow = (w + 8 * i) * 16 + l15;
      v16bf bhi = load_frag(WEh, nrow, kIN, kb);
      v16bf blo = load_frag(WEl, nrow, kIN, kb);
      acc[i] = wmma3(ahi, alo, bhi, blo, acc[i]);
    }
  }
  __syncthreads(); // all waves done reading the x tile before overwriting with E

#pragma unroll
  for (int i = 0; i < 4; ++i) {
    const int col = (w + 8 * i) * 16 + l15;
    const float bias = b_emb[c * kH + col];
#pragma unroll
    for (int j = 0; j < 8; ++j) {
      const int m = j + 8 * half;
      float e = acc[i][j] + bias;
      __bf16 h = (__bf16)e;
      Shi[m * kH + col] = h;
      Slo[m * kH + col] = (__bf16)(e - (float)h);
    }
  }
  __syncthreads();

  // ---- GEMM2: XG (16 x 1536), 96 col-tiles, 12 per wave ----
  v8f acc2[12];
#pragma unroll
  for (int i = 0; i < 12; ++i) acc2[i] = zero8();
  for (int k = 0; k < kH / 32; ++k) {
    const int kb = half * 8 + k * 32;
    v16bf ahi = load_frag(Shi, l15, kH, kb);
    v16bf alo = load_frag(Slo, l15, kH, kb);
#pragma unroll
    for (int i = 0; i < 12; ++i) {
      const int grow = (w + 8 * i) * 16 + l15;
      v16bf bhi = load_frag(WIh, grow, kH, kb);
      v16bf blo = load_frag(WIl, grow, kH, kb);
      acc2[i] = wmma3(ahi, alo, bhi, blo, acc2[i]);
    }
  }
#pragma unroll
  for (int i = 0; i < 12; ++i) {
    const int g = (w + 8 * i) * 16 + l15;
    const float bias = b_ih[c * kG + g];
#pragma unroll
    for (int j = 0; j < 8; ++j) {
      const int m = j + 8 * half;
      xg[(size_t)(b * kT + t0 + m) * kG + g] = acc2[i][j] + bias;
    }
  }
}

// ---------------- Kernel 4: GRU scan, WG per 16-sample camera tile ----------------
__global__ __launch_bounds__(256) void scan_kernel(
    const int* __restrict__ tile_cam, const int* __restrict__ tile_rows,
    const int* __restrict__ n_tiles,
    const bf16_t* __restrict__ Whh_hi, const bf16_t* __restrict__ Whh_lo,
    const float* __restrict__ b_hh, const float* __restrict__ xg,
    float* __restrict__ out) {
  __shared__ __align__(16) bf16_t Hhi[16 * kH];
  __shared__ __align__(16) bf16_t Hlo[16 * kH];

  const int tile = blockIdx.x;
  if (tile >= *n_tiles) return; // uniform across the block

  const int c = tile_cam[tile];
  const int w = threadIdx.x >> 5;
  const int lane = threadIdx.x & 31;
  const int half = lane >> 4;
  const int l15 = lane & 15;

  for (int i = threadIdx.x; i < 16 * kH; i += 256) {
    Hhi[i] = (__bf16)0.0f;
    Hlo[i] = (__bf16)0.0f;
  }

  int smp[8]; bool val[8];
#pragma unroll
  for (int j = 0; j < 8; ++j) {
    const int m = j + 8 * half;
    const int s = tile_rows[tile * 16 + m];
    val[j] = (s >= 0);
    smp[j] = val[j] ? s : 0;
  }

  int colc[4]; float bhr[4], bhz[4], bhn[4];
#pragma unroll
  for (int i = 0; i < 4; ++i) {
    const int col = (w + 8 * i) * 16 + l15;
    colc[i] = col;
    bhr[i] = b_hh[c * kG + col];
    bhz[i] = b_hh[c * kG + kH + col];
    bhn[i] = b_hh[c * kG + 2 * kH + col];
  }

  const bf16_t* WHh = Whh_hi + (size_t)c * kG * kH;
  const bf16_t* WHl = Whh_lo + (size_t)c * kG * kH;

  v8f hreg[4];
#pragma unroll
  for (int i = 0; i < 4; ++i) hreg[i] = zero8();

  for (int t = 0; t < kT; ++t) {
    __syncthreads(); // prior step's H hi/lo writes visible

    v8f ar[4], az[4], an[4];
#pragma unroll
    for (int i = 0; i < 4; ++i) { ar[i] = zero8(); az[i] = zero8(); an[i] = zero8(); }

    for (int k = 0; k < kH / 32; ++k) {
      const int kb = half * 8 + k * 32;
      v16bf ahi = load_frag(Hhi, l15, kH, kb);
      v16bf alo = load_frag(Hlo, l15, kH, kb);
#pragma unroll
      for (int i = 0; i < 4; ++i) {
        const int g = colc[i];
        { v16bf bh = load_frag(WHh, g,          kH, kb), bl = load_frag(WHl, g,          kH, kb); ar[i] = wmma3(ahi, alo, bh, bl, ar[i]); }
        { v16bf bh = load_frag(WHh, g + kH,     kH, kb), bl = load_frag(WHl, g + kH,     kH, kb); az[i] = wmma3(ahi, alo, bh, bl, az[i]); }
        { v16bf bh = load_frag(WHh, g + 2 * kH, kH, kb), bl = load_frag(WHl, g + 2 * kH, kH, kb); an[i] = wmma3(ahi, alo, bh, bl, an[i]); }
      }
    }
    __syncthreads(); // all waves finished reading H before overwrite

#pragma unroll
    for (int i = 0; i < 4; ++i) {
      const int col = colc[i];
#pragma unroll
      for (int j = 0; j < 8; ++j) {
        const int m = j + 8 * half;
        const size_t xb = (size_t)(smp[j] * kT + t) * kG + col;
        const float xr = xg[xb], xz = xg[xb + kH], xn = xg[xb + 2 * kH];
        // prefetch next step's xg values (speculative; dropped if unmapped)
        __builtin_prefetch(xg + xb + kG, 0, 0);
        __builtin_prefetch(xg + xb + kG + kH, 0, 0);
        __builtin_prefetch(xg + xb + kG + 2 * kH, 0, 0);
        const float hr = ar[i][j] + bhr[i];
        const float hz = az[i][j] + bhz[i];
        const float hn = an[i][j] + bhn[i];
        const float r = 1.0f / (1.0f + __expf(-(xr + hr)));
        const float z = 1.0f / (1.0f + __expf(-(xz + hz)));
        const float nn = tanhf(xn + r * hn);
        const float hnew = (1.0f - z) * nn + z * hreg[i][j];
        hreg[i][j] = hnew;
        __bf16 hh = (__bf16)hnew;
        Hhi[m * kH + col] = hh;
        Hlo[m * kH + col] = (__bf16)(hnew - (float)hh);
      }
    }
  }

#pragma unroll
  for (int i = 0; i < 4; ++i) {
    const int col = colc[i];
#pragma unroll
    for (int j = 0; j < 8; ++j) {
      if (val[j]) out[(size_t)smp[j] * kH + col] = hreg[i][j];
    }
  }
}

extern "C" void kernel_launch(void* const* d_in, const int* in_sizes, int n_in,
                              void* d_out, int out_size, void* d_ws, size_t ws_size,
                              hipStream_t stream) {
  (void)in_sizes; (void)n_in; (void)out_size; (void)ws_size;

  const float* x     = (const float*)d_in[0];
  const int*   cam   = (const int*)d_in[1];
  const float* W_emb = (const float*)d_in[2];
  const float* b_emb = (const float*)d_in[3];
  const float* W_ih  = (const float*)d_in[4];
  const float* b_ih  = (const float*)d_in[5];
  const float* W_hh  = (const float*)d_in[6];
  const float* b_hh  = (const float*)d_in[7];
  float* out = (float*)d_out;

  char* ws = (char*)d_ws;
  size_t off = 0;
  auto alloc = [&](size_t bytes) {
    char* p = ws + off;
    off = (off + bytes + 255) & ~(size_t)255;
    return p;
  };
  bf16_t* Wemb_hi = (bf16_t*)alloc((size_t)nWemb * sizeof(bf16_t));
  bf16_t* Wemb_lo = (bf16_t*)alloc((size_t)nWemb * sizeof(bf16_t));
  bf16_t* Wih_hi  = (bf16_t*)alloc((size_t)nWg * sizeof(bf16_t));
  bf16_t* Wih_lo  = (bf16_t*)alloc((size_t)nWg * sizeof(bf16_t));
  bf16_t* Whh_hi  = (bf16_t*)alloc((size_t)nWg * sizeof(bf16_t));
  bf16_t* Whh_lo  = (bf16_t*)alloc((size_t)nWg * sizeof(bf16_t));
  bf16_t* x_hi    = (bf16_t*)alloc((size_t)nX * sizeof(bf16_t));
  bf16_t* x_lo    = (bf16_t*)alloc((size_t)nX * sizeof(bf16_t));
  float*  xg      = (float*)alloc((size_t)nXG * sizeof(float));
  int* tile_cam   = (int*)alloc(MAX_TILES * sizeof(int));
  int* tile_rows  = (int*)alloc(MAX_TILES * 16 * sizeof(int));
  int* n_tiles    = (int*)alloc(sizeof(int));

  split_kernel<<<(nWemb + 255) / 256, 256, 0, stream>>>(W_emb, Wemb_hi, Wemb_lo, nWemb);
  split_kernel<<<(nWg + 255) / 256, 256, 0, stream>>>(W_ih, Wih_hi, Wih_lo, nWg);
  split_kernel<<<(nWg + 255) / 256, 256, 0, stream>>>(W_hh, Whh_hi, Whh_lo, nWg);
  split_kernel<<<(nX + 255) / 256, 256, 0, stream>>>(x, x_hi, x_lo, nX);
  build_tiles_kernel<<<1, 1, 0, stream>>>(cam, tile_cam, tile_rows, n_tiles);
  emb_xg_kernel<<<kB * (kT / 16), 256, 0, stream>>>(x_hi, x_lo, cam, Wemb_hi, Wemb_lo, b_emb,
                                                    Wih_hi, Wih_lo, b_ih, xg);
  scan_kernel<<<MAX_TILES, 256, 0, stream>>>(tile_cam, tile_rows, n_tiles,
                                             Whh_hi, Whh_lo, b_hh, xg, out);
}
